// GATLayer_10840497455591
// MI455X (gfx1250) — matrix operand order repaired and checked by reading, not compiled, not run on hardware
//
#include <hip/hip_runtime.h>
#include <math.h>

typedef _Float16 h16;
typedef __attribute__((ext_vector_type(16))) _Float16 v16h;
typedef __attribute__((ext_vector_type(8)))  _Float16 v8h;
typedef __attribute__((ext_vector_type(8)))  float    v8f;

#define NN   8192
#define KIN  512
#define FOUT 256
#define EHC  288       // cols 0..255 = e*H, col 256 = e, 257..287 = zero pad (18 N-tiles)
#define APAD 40        // LDS A-tile pitch in halves: 80B rows -> 16B aligned + conflict-free

union V16U { v16h v; v8h h[2]; };

// A-fragment (16x32 f16 tile, row-major in LDS, pitch in halves), contiguous K to match B:
// lane<16: elems 0..15 -> K 0..15 (row = lane)
// lane>=16: elems 0..15 -> K 16..31 (row = lane-16)
__device__ __forceinline__ v16h load_a_frag(const h16* base, int pitch, int lane) {
  const int row = lane & 15, hs = lane >> 4;
  V16U u;
  u.h[0] = *(const v8h*)(base + row * pitch + hs * 16);
  u.h[1] = *(const v8h*)(base + row * pitch + hs * 16 + 8);
  return u.v;
}

// B-fragment (32x16) loaded straight from a TRANSPOSED global matrix T[col][k]:
// lane needs T[cb + (lane&15)][k0 + (lane>>4)*16 + e], e=0..15  -> 2x b128, contiguous.
__device__ __forceinline__ v16h load_b_frag_T(const h16* T, size_t pitch, int cb, int k0, int lane) {
  const int n = lane & 15, kh = lane >> 4;
  const h16* p = T + (size_t)(cb + n) * pitch + k0 + kh * 16;
  V16U u;
  u.h[0] = *(const v8h*)p;
  u.h[1] = *(const v8h*)(p + 8);
  return u.v;
}

// ---- Kernel 0: Xh = f16(X); WhT = f16(W)^T  (tiny one-time conversions) ----
__global__ void __launch_bounds__(256) k_convert(const float* __restrict__ X,
                                                 const float* __restrict__ W,
                                                 h16* __restrict__ Xh, h16* __restrict__ WhT) {
  const int stride = gridDim.x * blockDim.x;
  for (int i = blockIdx.x * blockDim.x + threadIdx.x; i < NN * KIN; i += stride)
    Xh[i] = (h16)X[i];
  for (int i = blockIdx.x * blockDim.x + threadIdx.x; i < KIN * FOUT; i += stride) {
    const int c = i / KIN, k = i % KIN;          // WhT[c][k] = W[k][c]
    WhT[i] = (h16)W[(size_t)k * FOUT + c];
  }
}

// ---- Kernel 1: H = Xh @ W + bias. Block: 32 rows x 256 cols, 8 waves, 4 WMMA/wave/iter ----
__global__ void __launch_bounds__(256) k_gemm_h(const h16* __restrict__ Xh,
                                                const h16* __restrict__ WhT,
                                                const float* __restrict__ bias,
                                                float* __restrict__ H) {
  __shared__ __align__(16) h16 Xs[2][32][APAD];
  const int tid = threadIdx.x;
  const int wave = tid >> 5, lane = tid & 31;
  const int i0 = blockIdx.x * 32;
  const int cb0 = wave * 32, cb1 = cb0 + 16;
  v8f c00 = {}, c01 = {}, c10 = {}, c11 = {};
  int buf = 0;
  for (int k0 = 0; k0 < KIN; k0 += 32, buf ^= 1) {
    {  // stage 32x32 f16 of X, 8B per thread, coalesced
      const int r = tid >> 3, s = (tid & 7) * 4;
      *(uint2*)&Xs[buf][r][s] = *(const uint2*)&Xh[(size_t)(i0 + r) * KIN + k0 + s];
    }
    __syncthreads();
    const v16h a0 = load_a_frag(&Xs[buf][0][0], APAD, lane);
    const v16h a1 = load_a_frag(&Xs[buf][16][0], APAD, lane);
    const v16h b0 = load_b_frag_T(WhT, KIN, cb0, k0, lane);
    const v16h b1 = load_b_frag_T(WhT, KIN, cb1, k0, lane);
    c00 = __builtin_amdgcn_wmma_f32_16x16x32_f16(false, a0, false, b0, (short)0, c00, false, false);
    c01 = __builtin_amdgcn_wmma_f32_16x16x32_f16(false, a0, false, b1, (short)0, c01, false, false);
    c10 = __builtin_amdgcn_wmma_f32_16x16x32_f16(false, a1, false, b0, (short)0, c10, false, false);
    c11 = __builtin_amdgcn_wmma_f32_16x16x32_f16(false, a1, false, b1, (short)0, c11, false, false);
  }
  const int n = lane & 15, hs = lane >> 4;
  const float bv0 = bias[cb0 + n], bv1 = bias[cb1 + n];
#pragma unroll
  for (int r = 0; r < 8; ++r) {
    const int row0 = i0 + r + 8 * hs, row1 = row0 + 16;
    H[(size_t)row0 * FOUT + cb0 + n] = c00[r] + bv0;
    H[(size_t)row0 * FOUT + cb1 + n] = c01[r] + bv1;
    H[(size_t)row1 * FOUT + cb0 + n] = c10[r] + bv0;
    H[(size_t)row1 * FOUT + cb1 + n] = c11[r] + bv1;
  }
}

// ---- Kernel 2: s_dst[i] = H[i,:] . phi[FOUT:] (one wave per row) ----
__global__ void __launch_bounds__(256) k_sdst(const float* __restrict__ H,
                                              const float* __restrict__ phi,
                                              float* __restrict__ sdst) {
  const int wave = threadIdx.x >> 5, lane = threadIdx.x & 31;
  const int row = blockIdx.x * 8 + wave;
  float acc = 0.f;
#pragma unroll
  for (int q = 0; q < FOUT / 32; ++q) {
    const int f = lane + q * 32;
    acc += H[(size_t)row * FOUT + f] * phi[FOUT + f];
  }
#pragma unroll
  for (int m = 16; m >= 1; m >>= 1) acc += __shfl_xor(acc, m, 32);
  if (lane == 0) sdst[row] = acc;
}

// ---- Kernel 3: m = max_i s_dst[i] ----
__global__ void __launch_bounds__(1024) k_max(const float* __restrict__ sdst,
                                              float* __restrict__ mout) {
  __shared__ float red[1024];
  float m = -1e30f;
  for (int i = threadIdx.x; i < NN; i += 1024) m = fmaxf(m, sdst[i]);
  red[threadIdx.x] = m;
  __syncthreads();
  for (int s = 512; s >= 1; s >>= 1) {
    if ((int)threadIdx.x < s) red[threadIdx.x] = fmaxf(red[threadIdx.x], red[threadIdx.x + s]);
    __syncthreads();
  }
  if (threadIdx.x == 0) mout[0] = red[0];
}

// ---- Kernel 4: EhT[c][k] = transposed [e*H | e | 0] in f16 (LDS tile transpose) ----
__global__ void __launch_bounds__(1024) k_build_ehT(const float* __restrict__ H,
                                                    const float* __restrict__ sdst,
                                                    const float* __restrict__ mval,
                                                    h16* __restrict__ EhT) {
  __shared__ h16 tile[32][33];
  const int k0 = blockIdx.x * 32;                 // source rows (k of H)
  const int c0 = blockIdx.y * 32;                 // columns
  const int tx = threadIdx.x & 31, ty = threadIdx.x >> 5;
  const int k = k0 + ty, c = c0 + tx;
  const float e = __expf(sdst[k] - mval[0]);
  float v = 0.f;
  if (c < FOUT) v = e * H[(size_t)k * FOUT + c];
  else if (c == FOUT) v = e;
  tile[ty][tx] = (h16)v;
  __syncthreads();
  EhT[(size_t)(c0 + ty) * NN + k0 + tx] = tile[tx][ty];   // coalesced along k
}

// ---- Kernel 5: num = Mask @ [e*H | e], out = num[:, :256] / num[:, 256] ----
// Block: 32 rows x 288 cols, 9 waves, 4 WMMA/wave/iter; adj read once, coalesced.
__global__ void __launch_bounds__(288) k_attn(const int* __restrict__ adj,
                                              const h16* __restrict__ EhT,
                                              float* __restrict__ out) {
  __shared__ __align__(16) h16 Ms[2][32][APAD];
  __shared__ float denom[32];
  const int tid = threadIdx.x;
  const int wave = tid >> 5, lane = tid & 31;
  const int i0 = blockIdx.x * 32;
  const int cb0 = wave * 32, cb1 = cb0 + 16;
  v8f c00 = {}, c01 = {}, c10 = {}, c11 = {};
  int buf = 0;
  for (int k0 = 0; k0 < NN; k0 += 32, buf ^= 1) {
    if (tid < 256) {   // stage 32x32 mask tile: int4 load + pack 4 f16, coalesced
      const int r = tid >> 3, kk = (tid & 7) * 4;
      const int gi = i0 + r, gk = k0 + kk;
      const int4 av = *(const int4*)&adj[(size_t)gi * NN + gk];
      union { h16 h[4]; uint2 u; } p;
      p.h[0] = (h16)((av.x != 0 || gi == gk + 0) ? 1.0f : 0.0f);
      p.h[1] = (h16)((av.y != 0 || gi == gk + 1) ? 1.0f : 0.0f);
      p.h[2] = (h16)((av.z != 0 || gi == gk + 2) ? 1.0f : 0.0f);
      p.h[3] = (h16)((av.w != 0 || gi == gk + 3) ? 1.0f : 0.0f);
      *(uint2*)&Ms[buf][r][kk] = p.u;
    }
    if (k0 + 32 < NN && tid < 32)   // prefetch next adj slab (global_prefetch path)
      __builtin_prefetch(&adj[(size_t)(i0 + tid) * NN + k0 + 32], 0, 0);
    __syncthreads();
    const v16h a0 = load_a_frag(&Ms[buf][0][0], APAD, lane);
    const v16h a1 = load_a_frag(&Ms[buf][16][0], APAD, lane);
    const v16h b0 = load_b_frag_T(EhT, NN, cb0, k0, lane);
    const v16h b1 = load_b_frag_T(EhT, NN, cb1, k0, lane);
    c00 = __builtin_amdgcn_wmma_f32_16x16x32_f16(false, a0, false, b0, (short)0, c00, false, false);
    c01 = __builtin_amdgcn_wmma_f32_16x16x32_f16(false, a0, false, b1, (short)0, c01, false, false);
    c10 = __builtin_amdgcn_wmma_f32_16x16x32_f16(false, a1, false, b0, (short)0, c10, false, false);
    c11 = __builtin_amdgcn_wmma_f32_16x16x32_f16(false, a1, false, b1, (short)0, c11, false, false);
  }
  const int n = lane & 15, hs = lane >> 4;
  if (wave == 8 && n == 0) {        // wave 8, tile cb0: N=0 is global col 256 = denominator
#pragma unroll
    for (int r = 0; r < 8; ++r) {
      denom[r + 8 * hs]      = c00[r];
      denom[16 + r + 8 * hs] = c10[r];
    }
  }
  __syncthreads();
  if (wave < 8) {
#pragma unroll
    for (int r = 0; r < 8; ++r) {
      const int m0 = r + 8 * hs, m1 = m0 + 16;
      const int row0 = i0 + m0, row1 = i0 + m1;
      const float d0 = denom[m0], d1 = denom[m1];
      out[(size_t)row0 * FOUT + cb0 + n] = c00[r] / d0;
      out[(size_t)row0 * FOUT + cb1 + n] = c01[r] / d0;
      out[(size_t)row1 * FOUT + cb0 + n] = c10[r] / d1;
      out[(size_t)row1 * FOUT + cb1 + n] = c11[r] / d1;
    }
  }
}

extern "C" void kernel_launch(void* const* d_in, const int* in_sizes, int n_in,
                              void* d_out, int out_size, void* d_ws, size_t ws_size,
                              hipStream_t stream) {
  const float* X    = (const float*)d_in[0];
  const int*   adj  = (const int*)d_in[1];
  const float* W    = (const float*)d_in[2];
  const float* bias = (const float*)d_in[3];
  const float* phi  = (const float*)d_in[4];
  float* out = (float*)d_out;

  char* ws = (char*)d_ws;
  size_t off = 0;
  h16*   Xh   = (h16*)(ws + off);   off += (size_t)NN * KIN * 2;    // 8 MB
  h16*   WhT  = (h16*)(ws + off);   off += (size_t)KIN * FOUT * 2;  // 256 KB
  float* H    = (float*)(ws + off); off += (size_t)NN * FOUT * 4;   // 8 MB
  h16*   EhT  = (h16*)(ws + off);   off += (size_t)EHC * NN * 2;    // 4.5 MB
  float* sdst = (float*)(ws + off); off += (size_t)NN * 4;          // 32 KB
  float* mval = (float*)(ws + off); off += 16;

  k_convert  <<<dim3(1024),        dim3(256),  0, stream>>>(X, W, Xh, WhT);
  k_gemm_h   <<<dim3(NN / 32),     dim3(256),  0, stream>>>(Xh, WhT, bias, H);
  k_sdst     <<<dim3(NN / 8),      dim3(256),  0, stream>>>(H, phi, sdst);
  k_max      <<<dim3(1),           dim3(1024), 0, stream>>>(sdst, mval);
  k_build_ehT<<<dim3(NN / 32, 9),  dim3(1024), 0, stream>>>(H, sdst, mval, EhT);
  k_attn     <<<dim3(NN / 32),     dim3(288),  0, stream>>>(adj, EhT, out);
}
